// ElementWiseGating_47863115547496
// MI455X (gfx1250) — compile-verified, matrix operand order checked by reading, not run on hardware
//
#include <hip/hip_runtime.h>

typedef __attribute__((ext_vector_type(16))) __bf16 v16bf;
typedef __attribute__((ext_vector_type(8)))  float  v8f;
typedef unsigned int u32x4 __attribute__((ext_vector_type(4)));
typedef int          i32x8 __attribute__((ext_vector_type(8)));
typedef int          i32x4 __attribute__((ext_vector_type(4)));

#if __has_builtin(__builtin_amdgcn_tensor_load_to_lds)
#define HAS_TDM 1
#else
#define HAS_TDM 0
#endif

#define LEAKY 0.01f

// problem sizes
#define Bsz 65536
#define Lsz 256
#define Osz 256
#define Esz 8
#define H1  128
#define H2  64
#define MT  64          // batch rows per block
#define NTHREADS 256

// ---- LDS layout (bytes) ----
#define XS_OFF   0               // x tile bf16 [64][264]
#define XS_STR   264
#define H2S_OFF  33792           // h2 bf16 [8][64][72]
#define H2S_STR  72
#define WA_OFF   107520          // staged B-weights bf16 [128][136]; also combine scratch
#define WA_STR   136
#define WB_OFF   142336          // h1s bf16 [64][136]  OR  w3t bf16 [128][72]
#define H1S_STR  136
#define W3T_STR  72
#define B1S_OFF  160768
#define B2S_OFF  161280
#define B3T_OFF  161536
#define BGT_OFF  162048
#define SMEM_BYTES 162560

// ---- workspace layout (bytes): pre-converted bf16, pre-transposed, pre-padded ----
#define WS_XBF   0ull                 // [65536][256] bf16               = 33,554,432
#define WS_W1T   33554432ull          // [8][2][128][136] bf16           =    557,056
#define WS_W2T   34111488ull          // [8][64][136] bf16               =    139,264
#define WS_W3T   34250752ull          // [16][128][72] bf16              =    294,912
#define WS_WGT   34545664ull          // [16][2][128][136] bf16          =  1,114,112
#define WS_NEEDED 35659776ull

__device__ __forceinline__ unsigned short f2bf(float f) {
  unsigned int u = __float_as_uint(f);
  return (unsigned short)((u + 0x7FFFu + ((u >> 16) & 1u)) >> 16);
}

__device__ __forceinline__ v8f vzero8() {
  v8f z;
#pragma unroll
  for (int i = 0; i < 8; ++i) z[i] = 0.0f;
  return z;
}

// 16x32 bf16 WMMA fragment from LDS; major dim (M for A / N for B) has `stride`
// bf16 elements, K contiguous. Two ds_load_b128 per lane per ISA A/B layout.
__device__ __forceinline__ v16bf ldsFrag(const unsigned short* base, int stride,
                                         int rc0, int k0, int lane) {
  const int hh = lane >> 4;
  const int r  = lane & 15;
  const unsigned short* p = base + (rc0 + r) * stride + k0 + hh * 8;
  union { uint4 q[2]; v16bf v; } u;
  u.q[0] = *(const uint4*)(p);
  u.q[1] = *(const uint4*)(p + 16);
  return u.v;
}

// ---- Tensor Data Mover: 2D tile (global bf16 -> LDS), D# per CDNA5 ISA §8 ----
// tile0 = elements per row, tile1 = rows, stride0 = row stride (elements),
// w0x = extra D# group1 word0 bits (pad ctrl). 1D copy: tile1=1, stride0=tile0.
__device__ __forceinline__ void tdm_load(unsigned lds_off, const void* g,
                                         unsigned tile0, unsigned tile1,
                                         unsigned stride0, unsigned w0x) {
#if HAS_TDM
  unsigned long long ga = (unsigned long long)g;
  u32x4 g0;
  g0.x = 1u;                                   // count=1, user-mode load
  g0.y = lds_off;                              // lds_addr (bytes)
  g0.z = (unsigned)ga;                         // global_addr[31:0]
  g0.w = (unsigned)((ga >> 32) & 0x1FFFFFFu) | (2u << 30);  // addr[56:32] | type=2
  unsigned td0 = stride0;                      // tensor_dim0 >= row length
  i32x8 g1;
  g1[0] = (int)((1u << 16) | w0x);             // data_size=1 (2B) | pad ctl
  g1[1] = (int)((td0 & 0xFFFFu) << 16);        // [47:32]=atomic_bar=0 | tensor_dim0.lo
  g1[2] = (int)(((td0 >> 16) & 0xFFFFu) | ((tile1 & 0xFFFFu) << 16)); // td0.hi | tensor_dim1.lo
  g1[3] = (int)(((tile1 >> 16) & 0xFFFFu) | ((tile0 & 0xFFFFu) << 16)); // td1.hi | tile_dim0
  g1[4] = (int)(tile1 & 0xFFFFu);              // tile_dim1 | tile_dim2=0
  g1[5] = (int)stride0;                        // tensor_dim0_stride.lo
  g1[6] = 0;                                   // stride0.hi | tensor_dim1_stride.lo
  g1[7] = 0;
  i32x4 z4 = {0, 0, 0, 0};
  i32x8 z8 = {0, 0, 0, 0, 0, 0, 0, 0};
  // clang-23 / therock signature: (g0, g1, g2, g3, g4, cpol)
  __builtin_amdgcn_tensor_load_to_lds(g0, g1, z4, z4, z8, 0);
#else
  (void)lds_off; (void)g; (void)tile0; (void)tile1; (void)stride0; (void)w0x;
#endif
}

__device__ __forceinline__ void wait_tensor_le1() {
#if __has_builtin(__builtin_amdgcn_s_wait_tensorcnt)
  __builtin_amdgcn_s_wait_tensorcnt(1);
#else
  asm volatile("s_wait_tensorcnt 0x1" ::: "memory");
#endif
}
__device__ __forceinline__ void wait_tensor_le0() {
#if __has_builtin(__builtin_amdgcn_s_wait_tensorcnt)
  __builtin_amdgcn_s_wait_tensorcnt(0);
#else
  asm volatile("s_wait_tensorcnt 0x0" ::: "memory");
#endif
}

// ===================== workspace prep kernels (run once per call) ============
__global__ void prep_x(const float* __restrict__ x, unsigned short* __restrict__ xbf) {
  int i = blockIdx.x * blockDim.x + threadIdx.x;           // 4 elems per thread
  const float4* x4 = (const float4*)x;
  float4 v = x4[i];
  ushort4 o;
  o.x = f2bf(v.x); o.y = f2bf(v.y); o.z = f2bf(v.z); o.w = f2bf(v.w);
  ((ushort4*)xbf)[i] = o;
}
__global__ void prep_w1t(const float* __restrict__ W1, unsigned short* __restrict__ dst) {
  int i = blockIdx.x * blockDim.x + threadIdx.x;           // 8*2*128*128
  int n = i & 127, kk = (i >> 7) & 127, kc = (i >> 14) & 1, e = i >> 15;
  dst[(size_t)((e * 2 + kc) * 128 + n) * 136 + kk] =
      f2bf(W1[(size_t)(e * Lsz + kc * 128 + kk) * H1 + n]);
}
__global__ void prep_w2t(const float* __restrict__ W2, unsigned short* __restrict__ dst) {
  int i = blockIdx.x * blockDim.x + threadIdx.x;           // 8*128*64
  int n = i & 63, kk = (i >> 6) & 127, e = i >> 13;
  dst[(size_t)(e * 64 + n) * 136 + kk] = f2bf(W2[(size_t)(e * H1 + kk) * H2 + n]);
}
__global__ void prep_w3t(const float* __restrict__ W3, unsigned short* __restrict__ dst) {
  int i = blockIdx.x * blockDim.x + threadIdx.x;           // 8*64*256
  int o = i & 255, kk = (i >> 8) & 63, e = i >> 14;
  int t = o >> 4, c = o & 15;
  dst[(size_t)(t * 128 + e * 16 + c) * 72 + kk] = f2bf(W3[(size_t)(e * H2 + kk) * Osz + o]);
}
__global__ void prep_wgt(const float* __restrict__ Wg, unsigned short* __restrict__ dst) {
  int i = blockIdx.x * blockDim.x + threadIdx.x;           // 2048*128*2
  int cf = i & 2047, kk = (i >> 11) & 127, kc = i >> 18;
  int e = cf >> 8, t = (cf >> 4) & 15, c = cf & 15;
  dst[(size_t)((t * 2 + kc) * 128 + e * 16 + c) * 136 + kk] =
      f2bf(Wg[(size_t)(kc * 128 + kk) * (Esz * Osz) + cf]);
}

// ============================ main fused kernel ==============================
template <bool TDM>
__global__ __launch_bounds__(NTHREADS, 1)
void moe_gated_kernel(const float* __restrict__ x,
                      const float* __restrict__ W1, const float* __restrict__ b1,
                      const float* __restrict__ W2, const float* __restrict__ b2,
                      const float* __restrict__ W3, const float* __restrict__ b3,
                      const float* __restrict__ Wg, const float* __restrict__ bg,
                      float* __restrict__ out,
                      const unsigned short* __restrict__ xbf,
                      const unsigned short* __restrict__ w1t,
                      const unsigned short* __restrict__ w2t,
                      const unsigned short* __restrict__ w3tw,
                      const unsigned short* __restrict__ wgtw) {
  extern __shared__ char smem[];
  unsigned short* xs  = (unsigned short*)(smem + XS_OFF);
  unsigned short* h2s = (unsigned short*)(smem + H2S_OFF);
  unsigned short* wa  = (unsigned short*)(smem + WA_OFF);
  unsigned short* wb  = (unsigned short*)(smem + WB_OFF);
  float* b1s  = (float*)(smem + B1S_OFF);
  float* b2s  = (float*)(smem + B2S_OFF);
  float* b3t  = (float*)(smem + B3T_OFF);
  float* bgt  = (float*)(smem + BGT_OFF);
  float* comb = (float*)(smem + WA_OFF);   // [strip][3][256] f32, 12 KB

  const int tid   = threadIdx.x;
  const int wave  = tid >> 5;
  const int lane  = tid & 31;
  const int strip = wave >> 1;
  const int half2 = wave & 1;
  const int row0  = blockIdx.x * MT;
  const int srow  = strip * 16;
  const int lh    = lane >> 4;
  const int ln    = lane & 15;

  // ---------------- stage x tile ----------------
  if constexpr (TDM) {
    // DMA with LDS padding: 4 DWORDs pad every 128 DWORDs -> 264-elem rows
    if (wave == 0) {
      tdm_load(XS_OFF, xbf + (size_t)row0 * Lsz, /*tile0=*/Lsz, /*tile1=*/MT,
               /*stride0=*/Lsz, /*pad:*/ (1u << 20) | (6u << 22) | (3u << 25));
      wait_tensor_le0();
    }
  } else {
    for (int i = tid; i < MT * Lsz; i += NTHREADS) {
      int r = i >> 8, k = i & 255;
      xs[r * XS_STR + k] = f2bf(x[(size_t)(row0 + r) * Lsz + k]);
    }
  }
  __syncthreads();

  // ================= Phase A : h2[e] for all experts =================
  for (int e = 0; e < Esz; ++e) {
    for (int i = tid; i < H1; i += NTHREADS) b1s[i] = b1[e * H1 + i];
    for (int i = tid; i < H2; i += NTHREADS) b2s[i] = b2[e * H2 + i];

    v8f acc1[4];
#pragma unroll
    for (int t = 0; t < 4; ++t) acc1[t] = vzero8();

    for (int kc = 0; kc < 2; ++kc) {
      if constexpr (TDM) {
        if (wave == 0) {
          tdm_load(WA_OFF, w1t + (size_t)(e * 2 + kc) * 128 * 136, 128 * 136, 1, 128 * 136, 0);
          wait_tensor_le0();
        }
      } else {
        for (int i = tid; i < 128 * 128; i += NTHREADS) {
          int n = i & 127, kk = i >> 7;
          wa[n * WA_STR + kk] = f2bf(W1[(size_t)(e * Lsz + kc * 128 + kk) * H1 + n]);
        }
      }
      __syncthreads();
#pragma unroll
      for (int nt = 0; nt < 4; ++nt) {
        int col0 = half2 * 64 + nt * 16;
#pragma unroll
        for (int kk = 0; kk < 4; ++kk) {
          v16bf a  = ldsFrag(xs, XS_STR, srow, kc * 128 + kk * 32, lane);
          v16bf bm = ldsFrag(wa, WA_STR, col0, kk * 32, lane);
          acc1[nt] = __builtin_amdgcn_wmma_f32_16x16x32_bf16(
              false, a, false, bm, (short)0, acc1[nt], false, false);
        }
      }
      __syncthreads();
    }
#pragma unroll
    for (int nt = 0; nt < 4; ++nt) {
      int col = half2 * 64 + nt * 16 + ln;
      float bias = b1s[col];
#pragma unroll
      for (int v = 0; v < 8; ++v) {
        float val = acc1[nt][v] + bias;
        val = (val >= 0.0f) ? val : LEAKY * val;
        wb[(srow + v + 8 * lh) * H1S_STR + col] = f2bf(val);
      }
    }
    __syncthreads();

    if constexpr (TDM) {
      if (wave == 0) {
        tdm_load(WA_OFF, w2t + (size_t)e * 64 * 136, 64 * 136, 1, 64 * 136, 0);
        wait_tensor_le0();
      }
    } else {
      for (int i = tid; i < H2 * H1; i += NTHREADS) {
        int n = i & 63, kk = i >> 6;
        wa[n * WA_STR + kk] = f2bf(W2[(size_t)(e * H1 + kk) * H2 + n]);
      }
    }
    __syncthreads();

    v8f acc2[2];
#pragma unroll
    for (int t = 0; t < 2; ++t) acc2[t] = vzero8();
#pragma unroll
    for (int nt = 0; nt < 2; ++nt) {
      int col0 = half2 * 32 + nt * 16;
#pragma unroll
      for (int kk = 0; kk < 4; ++kk) {
        v16bf a  = ldsFrag(wb, H1S_STR, srow, kk * 32, lane);
        v16bf bm = ldsFrag(wa, WA_STR, col0, kk * 32, lane);
        acc2[nt] = __builtin_amdgcn_wmma_f32_16x16x32_bf16(
            false, a, false, bm, (short)0, acc2[nt], false, false);
      }
    }
#pragma unroll
    for (int nt = 0; nt < 2; ++nt) {
      int col = half2 * 32 + nt * 16 + ln;
      float bias = b2s[col];
#pragma unroll
      for (int v = 0; v < 8; ++v) {
        float val = acc2[nt][v] + bias;
        val = (val >= 0.0f) ? val : LEAKY * val;
        h2s[(e * MT + srow + v + 8 * lh) * H2S_STR + col] = f2bf(val);
      }
    }
    __syncthreads();
  }

  // ================= Phase B : gate + expert_out + softmax-combine ========
  const int eBase = half2 * 4;
  for (int n0 = 0; n0 < Osz; n0 += 16) {
    const int t16 = n0 >> 4;
    if constexpr (TDM) {
      if (wave == 0) {   // W3 tile -> wb, Wg chunk0 -> wa (pipelined behind eo)
        tdm_load(WB_OFF, w3tw + (size_t)t16 * 128 * 72, 128 * 72, 1, 128 * 72, 0);
        tdm_load(WA_OFF, wgtw + (size_t)(t16 * 2 + 0) * 128 * 136, 128 * 136, 1, 128 * 136, 0);
      }
    } else {
      for (int i = tid; i < Esz * 16 * H2; i += NTHREADS) {
        int c = i & 15, ee = (i >> 4) & 7, kk = i >> 7;
        wb[(ee * 16 + c) * W3T_STR + kk] = f2bf(W3[(size_t)(ee * H2 + kk) * Osz + n0 + c]);
      }
    }
    for (int i = tid; i < 128; i += NTHREADS) {
      int c = i & 15, ee = i >> 4;
      b3t[i] = b3[ee * Osz + n0 + c];
      bgt[i] = bg[ee * Osz + n0 + c];
    }
    if constexpr (TDM) { if (wave == 0) wait_tensor_le1(); }   // W3 landed
    __syncthreads();

    v8f eo[4];
#pragma unroll
    for (int j = 0; j < 4; ++j) {
      eo[j] = vzero8();
      int e = eBase + j;
#pragma unroll
      for (int kk = 0; kk < 2; ++kk) {
        v16bf a  = ldsFrag(h2s + e * MT * H2S_STR, H2S_STR, srow, kk * 32, lane);
        v16bf bm = ldsFrag(wb, W3T_STR, e * 16, kk * 32, lane);
        eo[j] = __builtin_amdgcn_wmma_f32_16x16x32_bf16(
            false, a, false, bm, (short)0, eo[j], false, false);
      }
    }

    v8f g[4];
#pragma unroll
    for (int j = 0; j < 4; ++j) g[j] = vzero8();
    for (int kc = 0; kc < 2; ++kc) {
      if constexpr (TDM) {
        if (kc == 1 && wave == 0)
          tdm_load(WA_OFF, wgtw + (size_t)(t16 * 2 + 1) * 128 * 136, 128 * 136, 1, 128 * 136, 0);
        if (wave == 0) wait_tensor_le0();
      } else {
        __syncthreads();
        for (int i = tid; i < Esz * 16 * 128; i += NTHREADS) {
          int c = i & 15, ee = (i >> 4) & 7, kk = i >> 7;
          wa[(ee * 16 + c) * WA_STR + kk] =
              f2bf(Wg[(size_t)(kc * 128 + kk) * (Esz * Osz) + ee * Osz + n0 + c]);
        }
      }
      __syncthreads();
#pragma unroll
      for (int j = 0; j < 4; ++j) {
        int e = eBase + j;
#pragma unroll
        for (int kk = 0; kk < 4; ++kk) {
          v16bf a  = ldsFrag(xs, XS_STR, srow, kc * 128 + kk * 32, lane);
          v16bf bm = ldsFrag(wa, WA_STR, e * 16, kk * 32, lane);
          g[j] = __builtin_amdgcn_wmma_f32_16x16x32_bf16(
              false, a, false, bm, (short)0, g[j], false, false);
        }
      }
      if constexpr (TDM) { if (kc == 0) __syncthreads(); }  // wa reads done before kc1 DMA
    }
    __syncthreads();

    float pm[8], ps[8], pv[8];
#pragma unroll
    for (int v = 0; v < 8; ++v) {
      float gb[4];
#pragma unroll
      for (int j = 0; j < 4; ++j) gb[j] = g[j][v] + bgt[(eBase + j) * 16 + ln];
      float mx = fmaxf(fmaxf(gb[0], gb[1]), fmaxf(gb[2], gb[3]));
      float s = 0.0f, vv = 0.0f;
#pragma unroll
      for (int j = 0; j < 4; ++j) {
        float w = __expf(gb[j] - mx);
        s += w;
        vv += w * (eo[j][v] + b3t[(eBase + j) * 16 + ln]);
      }
      pm[v] = mx; ps[v] = s; pv[v] = vv;
    }

    float* cb = comb + strip * 3 * 256;
    if (half2 == 1) {
#pragma unroll
      for (int v = 0; v < 8; ++v) {
        int elt = (v + 8 * lh) * 16 + ln;
        cb[elt]       = pm[v];
        cb[256 + elt] = ps[v];
        cb[512 + elt] = pv[v];
      }
    }
    __syncthreads();
    if (half2 == 0) {
#pragma unroll
      for (int v = 0; v < 8; ++v) {
        int m = v + 8 * lh;
        int elt = m * 16 + ln;
        float m1 = cb[elt], s1 = cb[256 + elt], v1 = cb[512 + elt];
        float M  = fmaxf(pm[v], m1);
        float a0 = __expf(pm[v] - M), a1 = __expf(m1 - M);
        out[(size_t)(row0 + srow + m) * Osz + n0 + ln] =
            (pv[v] * a0 + v1 * a1) / (ps[v] * a0 + s1 * a1);
      }
    }
    __syncthreads();
  }
}

extern "C" void kernel_launch(void* const* d_in, const int* in_sizes, int n_in,
                              void* d_out, int out_size, void* d_ws, size_t ws_size,
                              hipStream_t stream) {
  (void)in_sizes; (void)n_in; (void)out_size;
  const float* x  = (const float*)d_in[0];
  const float* W1 = (const float*)d_in[1];
  const float* b1 = (const float*)d_in[2];
  const float* W2 = (const float*)d_in[3];
  const float* b2 = (const float*)d_in[4];
  const float* W3 = (const float*)d_in[5];
  const float* b3 = (const float*)d_in[6];
  const float* Wg = (const float*)d_in[7];
  const float* bg = (const float*)d_in[8];
  float* out = (float*)d_out;

  unsigned short* xbf  = (unsigned short*)((char*)d_ws + WS_XBF);
  unsigned short* w1tp = (unsigned short*)((char*)d_ws + WS_W1T);
  unsigned short* w2tp = (unsigned short*)((char*)d_ws + WS_W2T);
  unsigned short* w3tp = (unsigned short*)((char*)d_ws + WS_W3T);
  unsigned short* wgtp = (unsigned short*)((char*)d_ws + WS_WGT);

  const bool use_tdm = HAS_TDM && (d_ws != nullptr) && (ws_size >= WS_NEEDED);
  dim3 block(NTHREADS), grid(Bsz / MT);

  if (use_tdm) {
    hipLaunchKernelGGL(prep_x,   dim3((Bsz * Lsz / 4) / NTHREADS), block, 0, stream, x, xbf);
    hipLaunchKernelGGL(prep_w1t, dim3((8 * 2 * 128 * 128) / NTHREADS), block, 0, stream, W1, w1tp);
    hipLaunchKernelGGL(prep_w2t, dim3((8 * 128 * 64) / NTHREADS), block, 0, stream, W2, w2tp);
    hipLaunchKernelGGL(prep_w3t, dim3((8 * 64 * 256) / NTHREADS), block, 0, stream, W3, w3tp);
    hipLaunchKernelGGL(prep_wgt, dim3((2048 * 128 * 2) / NTHREADS), block, 0, stream, Wg, wgtp);
    (void)hipFuncSetAttribute((const void*)(moe_gated_kernel<true>),
                              hipFuncAttributeMaxDynamicSharedMemorySize, SMEM_BYTES);
    hipLaunchKernelGGL((moe_gated_kernel<true>), grid, block, SMEM_BYTES, stream,
                       x, W1, b1, W2, b2, W3, b3, Wg, bg, out, xbf, w1tp, w2tp, w3tp, wgtp);
  } else {
    (void)hipFuncSetAttribute((const void*)(moe_gated_kernel<false>),
                              hipFuncAttributeMaxDynamicSharedMemorySize, SMEM_BYTES);
    hipLaunchKernelGGL((moe_gated_kernel<false>), grid, block, SMEM_BYTES, stream,
                       x, W1, b1, W2, b2, W3, b3, Wg, bg, out, xbf, w1tp, w2tp, w3tp, wgtp);
  }
}
